// PCdown_62792421868384
// MI455X (gfx1250) — matrix operand order repaired and checked by reading, not compiled, not run on hardware
//
#include <hip/hip_runtime.h>
#include <math.h>

// ---------------------------------------------------------------------------
// PCdown (PointNet++ set abstraction) for MI455X / gfx1250, wave32 + WMMA.
// Pipeline:
//   1) stats_x_partial + bn1_finalize : analytic BN1 stats from x moments
//   2) feats_kernel  [WMMA f16]       : x -> relu(BN(xW1)) W2 + b2   (feats)
//   3) fps_kernel                     : farthest point sampling (seq, per-B WG)
//   4) knn_pool_kernel                : fused sqdist + top-16 + group + maxpool
//   5) mlp3_kernel   [WMMA f16]       : pooled @ W3^T (pre-BN)
//   6) bn2_partial + bn2_finalize     : deterministic BN2 stats
//   7) final_mlp_kernel [WMMA f16]    : relu(BN2(t)) @ W4^T + b4 -> d_out
// Conv biases b1/b3 cancel inside training-mode BN and are never read.
// ---------------------------------------------------------------------------

typedef __attribute__((ext_vector_type(16))) _Float16 v16h;
typedef __attribute__((ext_vector_type(8)))  float    v8f;

#define BB   8
#define NN   4096
#define SS   2048
#define NPTS (BB*NN)          // 32768
#define NQ   (BB*SS)          // 16384
#define C1   32
#define C2   64
#define CC   67
#define KNB  16
#define EPSB 1e-5f

// workspace layout (float offsets)
#define OFF_PART1 0                        // 64 blocks * 9
#define OFF_E1S   640                      // 32
#define OFF_E1B   672                      // 32
#define OFF_PART2 1024                     // 128 blocks * 128
#define OFF_E2S   17600                    // 64
#define OFF_E2B   17664                    // 64
#define OFF_FEATS 20480                    // NPTS*C2   = 2097152
#define OFF_NXYZ  (OFF_FEATS + NPTS*C2)    // NQ*3      = 49152
#define OFF_POOL  (OFF_NXYZ + NQ*3)        // NQ*CC     = 1097728
#define OFF_T     (OFF_POOL + NQ*CC)       // NQ*C2     = 1048576
// total ~ 4.31M floats (~17.3 MB) of d_ws

// ---------------- WMMA fragment helpers (per CDNA5 ISA) ---------
// 16-bit A (16x32): lane half h, row m = lane&15; element i -> K =
//   (i>=8?16:0) + 8*h + (i&7)
__device__ __forceinline__ int a_kmap(int i, int half) {
  return ((i & 8) << 1) + half * 8 + (i & 7);
}

__device__ __forceinline__ v16h load_a_lds(const float* As, int ldk, int kbase,
                                           int lane) {
  int half = lane >> 4, m = lane & 15;
  v16h a;
#pragma unroll
  for (int i = 0; i < 16; ++i)
    a[i] = (_Float16)As[m * ldk + kbase + a_kmap(i, half)];
  return a;
}

// 16-bit B (32x16): lane half h holds K = a_kmap(i, h) for column n = lane&15.
// Weights stored [Cout][Cin]; B[k][n] = W[nbase+n][kbase+k], zero-pad k>=cin.
__device__ __forceinline__ v16h load_b_w(const float* W, int cin, int nbase,
                                         int kbase, int lane) {
  int half = lane >> 4, n = nbase + (lane & 15);
  v16h b;
#pragma unroll
  for (int i = 0; i < 16; ++i) {
    int k = kbase + a_kmap(i, half);
    b[i] = (_Float16)(k < cin ? W[n * cin + k] : 0.0f);
  }
  return b;
}

__device__ __forceinline__ v8f wmma_f16(v16h a, v16h b, v8f c) {
  return __builtin_amdgcn_wmma_f32_16x16x32_f16(false, a, false, b, (short)0,
                                                c, false, false);
}

// ---------------- 1) x moments (deterministic two-stage) -------------------
__global__ void stats_x_partial(const float* __restrict__ x,
                                float* __restrict__ part) {
  __shared__ float red[8][9];
  int tid = threadIdx.x, lane = tid & 31, wv = tid >> 5;
  float s[9] = {0, 0, 0, 0, 0, 0, 0, 0, 0};
  for (int p = blockIdx.x * blockDim.x + tid; p < NPTS;
       p += gridDim.x * blockDim.x) {
    float a = x[p * 3 + 0], b = x[p * 3 + 1], c = x[p * 3 + 2];
    s[0] += a; s[1] += b; s[2] += c;
    s[3] += a * a; s[4] += a * b; s[5] += a * c;
    s[6] += b * b; s[7] += b * c; s[8] += c * c;
  }
#pragma unroll
  for (int j = 0; j < 9; ++j)
    for (int off = 16; off; off >>= 1) s[j] += __shfl_xor(s[j], off, 32);
  if (lane == 0)
#pragma unroll
    for (int j = 0; j < 9; ++j) red[wv][j] = s[j];
  __syncthreads();
  if (tid < 9) {
    float acc = 0.f;
    for (int w = 0; w < 8; ++w) acc += red[w][tid];
    part[blockIdx.x * 9 + tid] = acc;
  }
}

// BN1 analytically: mean_c = W1[c]·mu ; var_c = W1[c]^T M W1[c] - mean^2
__global__ void bn1_finalize(const float* __restrict__ part,
                             const float* __restrict__ W1,
                             const float* __restrict__ g1,
                             const float* __restrict__ be1,
                             float* __restrict__ e1s, float* __restrict__ e1b) {
  __shared__ float st[9];
  int tid = threadIdx.x;
  if (tid < 9) {
    float acc = 0.f;
    for (int bk = 0; bk < 64; ++bk) acc += part[bk * 9 + tid];
    st[tid] = acc * (1.0f / (float)NPTS);
  }
  __syncthreads();
  if (tid < C1) {
    float w0 = W1[tid * 3], w1 = W1[tid * 3 + 1], w2 = W1[tid * 3 + 2];
    float mean = w0 * st[0] + w1 * st[1] + w2 * st[2];
    float e2 = w0 * w0 * st[3] + w1 * w1 * st[6] + w2 * w2 * st[8] +
               2.f * (w0 * w1 * st[4] + w0 * w2 * st[5] + w1 * w2 * st[7]);
    float var = e2 - mean * mean;
    float sc = g1[tid] * rsqrtf(var + EPSB);
    e1s[tid] = sc;
    e1b[tid] = be1[tid] - mean * sc;  // conv bias b1 cancels in BN
  }
}

// ---------------- 2) first MLP: x -> feats  (two chained WMMA GEMMs) -------
#define FW 8
__global__ void feats_kernel(const float* __restrict__ x,
                             const float* __restrict__ W1,
                             const float* __restrict__ e1s,
                             const float* __restrict__ e1b,
                             const float* __restrict__ W2,
                             const float* __restrict__ b2,
                             float* __restrict__ feats) {
  __shared__ float As[FW][16][32];
  __shared__ float Hs[FW][16][32];
  int lane = threadIdx.x & 31, wv = threadIdx.x >> 5;
  int pbase = (blockIdx.x * FW + wv) * 16;
  for (int i = lane; i < 16 * 32; i += 32) {
    int r = i >> 5, c = i & 31;
    As[wv][r][c] = (c < 3) ? x[(pbase + r) * 3 + c] : 0.0f;
  }
  __syncthreads();
  v16h a = load_a_lds(&As[wv][0][0], 32, 0, lane);
  v16h bw0 = load_b_w(W1, 3, 0, 0, lane);
  v16h bw1 = load_b_w(W1, 3, 16, 0, lane);
  v8f c0 = {}, c1 = {};
  c0 = wmma_f16(a, bw0, c0);
  c1 = wmma_f16(a, bw1, c1);
  int half = lane >> 4, nn0 = lane & 15;
  float s0 = e1s[nn0], t0 = e1b[nn0], s1 = e1s[nn0 + 16], t1 = e1b[nn0 + 16];
#pragma unroll
  for (int v = 0; v < 8; ++v) {
    int r = v + half * 8;
    Hs[wv][r][nn0]      = fmaxf(c0[v] * s0 + t0, 0.0f);
    Hs[wv][r][nn0 + 16] = fmaxf(c1[v] * s1 + t1, 0.0f);
  }
  __syncthreads();
  v16h a2 = load_a_lds(&Hs[wv][0][0], 32, 0, lane);
#pragma unroll
  for (int t = 0; t < 4; ++t) {
    v16h bw = load_b_w(W2, 32, t * 16, 0, lane);
    v8f c = {};
    c = wmma_f16(a2, bw, c);
    float bias = b2[t * 16 + nn0];
#pragma unroll
    for (int v = 0; v < 8; ++v)
      feats[(pbase + v + half * 8) * C2 + t * 16 + nn0] = c[v] + bias;
  }
}

// ---------------- 3) farthest point sampling (1 WG per batch) --------------
__global__ void fps_kernel(const float* __restrict__ x,
                           float* __restrict__ new_xyz) {
  __shared__ float dist[NN];
  __shared__ float wbest[16];
  __shared__ int   wbidx[16];
  __shared__ int   scur;
  int b = blockIdx.x, tid = threadIdx.x;
  int lane = tid & 31, wv = tid >> 5;
  const float* xb = x + (size_t)b * NN * 3;
  for (int n = tid; n < NN; n += 512) dist[n] = 3.4e38f;
  if (tid == 0) scur = 0;
  __syncthreads();
  for (int s = 0; s < SS; ++s) {
    int cur = scur;
    float cx = xb[cur * 3], cy = xb[cur * 3 + 1], cz = xb[cur * 3 + 2];
    if (tid == 0) {
      float* nz = new_xyz + ((size_t)b * SS + s) * 3;
      nz[0] = cx; nz[1] = cy; nz[2] = cz;
    }
    float bv = -1.0f;
    int bi = 0x7fffffff;
    for (int n = tid; n < NN; n += 512) {
      float dx = xb[n * 3] - cx, dy = xb[n * 3 + 1] - cy,
            dz = xb[n * 3 + 2] - cz;
      float d = dx * dx + dy * dy + dz * dz;
      float dm = fminf(dist[n], d);
      dist[n] = dm;
      if (dm > bv) { bv = dm; bi = n; }   // n increasing: first max wins
    }
    for (int off = 16; off; off >>= 1) {
      float ov = __shfl_xor(bv, off, 32);
      int   oi = __shfl_xor(bi, off, 32);
      if (ov > bv || (ov == bv && oi < bi)) { bv = ov; bi = oi; }
    }
    if (lane == 0) { wbest[wv] = bv; wbidx[wv] = bi; }
    __syncthreads();
    if (tid == 0) {
      float v = wbest[0];
      int i0 = wbidx[0];
      for (int w = 1; w < 16; ++w)
        if (wbest[w] > v || (wbest[w] == v && wbidx[w] < i0)) {
          v = wbest[w]; i0 = wbidx[w];
        }
      scur = i0;
    }
    __syncthreads();
  }
}

// ---------------- 4) fused kNN(16) + group + maxpool (1 wave / query) ------
#define KW 4
__global__ void knn_pool_kernel(const float* __restrict__ x,
                                const float* __restrict__ feats,
                                const float* __restrict__ new_xyz,
                                float* __restrict__ pooled) {
  __shared__ float dist[KW][NN];   // 64 KB
  int lane = threadIdx.x & 31, wv = threadIdx.x >> 5;
  int q = blockIdx.x * KW + wv;            // 0..NQ-1
  int b = q >> 11;
  const float* xb = x + (size_t)b * NN * 3;
  const float* fb = feats + (size_t)b * NN * C2;
  float cx = new_xyz[q * 3], cy = new_xyz[q * 3 + 1], cz = new_xyz[q * 3 + 2];
  float* D = dist[wv];
  for (int n = lane; n < NN; n += 32) {
    float dx = xb[n * 3] - cx, dy = xb[n * 3 + 1] - cy, dz = xb[n * 3 + 2] - cz;
    D[n] = dx * dx + dy * dy + dz * dz;
  }
  __syncthreads();
  int sel[KNB];
  for (int k = 0; k < KNB; ++k) {
    float bv = 3.4e38f;
    int bi = 0x7fffffff;
    for (int n = lane; n < NN; n += 32) {
      float v = D[n];
      if (v < bv || (v == bv && n < bi)) { bv = v; bi = n; }
    }
    for (int off = 16; off; off >>= 1) {
      float ov = __shfl_xor(bv, off, 32);
      int   oi = __shfl_xor(bi, off, 32);
      if (ov < bv || (ov == bv && oi < bi)) { bv = ov; bi = oi; }
    }
    sel[k] = bi;                     // identical on all lanes after butterfly
    if (lane == (bi & 31)) D[bi] = 3.4e38f;
  }
  // concat(grouped_xyz - center, grouped_feats) then max over K neighbors
  for (int c = lane; c < CC; c += 32) {
    float m = -3.4e38f;
    float ctr = (c == 0) ? cx : ((c == 1) ? cy : cz);
#pragma unroll
    for (int k = 0; k < KNB; ++k) {
      int id = sel[k];
      float v = (c < 3) ? (xb[id * 3 + c] - ctr) : fb[id * C2 + (c - 3)];
      m = fmaxf(m, v);
    }
    pooled[(size_t)q * CC + c] = m;
  }
}

// ---------------- 5) pooled @ W3^T (pre-BN), K padded 67->96 ---------------
#define MW 4
__global__ void mlp3_kernel(const float* __restrict__ pooled,
                            const float* __restrict__ W3,
                            float* __restrict__ t_out) {
  __shared__ float As[MW][16][96];
  int lane = threadIdx.x & 31, wv = threadIdx.x >> 5;
  int pbase = (blockIdx.x * MW + wv) * 16;
  for (int i = lane; i < 16 * 96; i += 32) {
    int r = i / 96, c = i - r * 96;
    As[wv][r][c] = (c < CC) ? pooled[(size_t)(pbase + r) * CC + c] : 0.0f;
  }
  __syncthreads();
  v16h a0 = load_a_lds(&As[wv][0][0], 96, 0, lane);
  v16h a1 = load_a_lds(&As[wv][0][0], 96, 32, lane);
  v16h a2 = load_a_lds(&As[wv][0][0], 96, 64, lane);
  int half = lane >> 4, nn0 = lane & 15;
#pragma unroll
  for (int t = 0; t < 4; ++t) {
    v8f c = {};
    c = wmma_f16(a0, load_b_w(W3, CC, t * 16, 0, lane), c);
    c = wmma_f16(a1, load_b_w(W3, CC, t * 16, 32, lane), c);
    c = wmma_f16(a2, load_b_w(W3, CC, t * 16, 64, lane), c);
#pragma unroll
    for (int v = 0; v < 8; ++v)
      t_out[(size_t)(pbase + v + half * 8) * C2 + t * 16 + nn0] = c[v];
  }
}

// ---------------- 6) BN2 stats (deterministic two-stage) -------------------
__global__ void bn2_partial(const float* __restrict__ t,
                            float* __restrict__ part) {
  __shared__ float redS[4][64], redQ[4][64];
  int tid = threadIdx.x;
  int c = tid & 63, g = tid >> 6;
  int row0 = blockIdx.x * 128;
  float s = 0.f, q = 0.f;
  for (int r = g; r < 128; r += 4) {
    float v = t[(size_t)(row0 + r) * C2 + c];
    s += v; q += v * v;
  }
  redS[g][c] = s; redQ[g][c] = q;
  __syncthreads();
  if (tid < 64) {
    float ss = redS[0][tid] + redS[1][tid] + redS[2][tid] + redS[3][tid];
    float qq = redQ[0][tid] + redQ[1][tid] + redQ[2][tid] + redQ[3][tid];
    part[blockIdx.x * 128 + tid] = ss;
    part[blockIdx.x * 128 + 64 + tid] = qq;
  }
}

__global__ void bn2_finalize(const float* __restrict__ part,
                             const float* __restrict__ g2,
                             const float* __restrict__ be2,
                             float* __restrict__ e2s, float* __restrict__ e2b) {
  int c = threadIdx.x;
  if (c >= C2) return;
  float s = 0.f, q = 0.f;
  for (int bk = 0; bk < 128; ++bk) {
    s += part[bk * 128 + c];
    q += part[bk * 128 + 64 + c];
  }
  float inv = 1.0f / (float)NQ;
  float mean = s * inv;
  float var = q * inv - mean * mean;
  float sc = g2[c] * rsqrtf(var + EPSB);
  e2s[c] = sc;
  e2b[c] = be2[c] - mean * sc;       // conv bias b3 cancels in BN
}

// ---------------- 7) relu(BN2(t)) @ W4^T + b4 -> out -----------------------
__global__ void final_mlp_kernel(const float* __restrict__ t,
                                 const float* __restrict__ e2s,
                                 const float* __restrict__ e2b,
                                 const float* __restrict__ W4,
                                 const float* __restrict__ b4,
                                 float* __restrict__ out) {
  __shared__ float Hs[MW][16][64];
  int lane = threadIdx.x & 31, wv = threadIdx.x >> 5;
  int pbase = (blockIdx.x * MW + wv) * 16;
  for (int i = lane; i < 16 * 64; i += 32) {
    int r = i >> 6, c = i & 63;
    Hs[wv][r][c] =
        fmaxf(t[(size_t)(pbase + r) * C2 + c] * e2s[c] + e2b[c], 0.0f);
  }
  __syncthreads();
  v16h a0 = load_a_lds(&Hs[wv][0][0], 64, 0, lane);
  v16h a1 = load_a_lds(&Hs[wv][0][0], 64, 32, lane);
  int half = lane >> 4, nn0 = lane & 15;
#pragma unroll
  for (int tt = 0; tt < 4; ++tt) {
    v8f c = {};
    c = wmma_f16(a0, load_b_w(W4, C2, tt * 16, 0, lane), c);
    c = wmma_f16(a1, load_b_w(W4, C2, tt * 16, 32, lane), c);
    float bias = b4[tt * 16 + nn0];
#pragma unroll
    for (int v = 0; v < 8; ++v)
      out[(size_t)(pbase + v + half * 8) * C2 + tt * 16 + nn0] = c[v] + bias;
  }
}

// ---------------------------------------------------------------------------
extern "C" void kernel_launch(void* const* d_in, const int* in_sizes, int n_in,
                              void* d_out, int out_size, void* d_ws,
                              size_t ws_size, hipStream_t stream) {
  (void)in_sizes; (void)n_in; (void)out_size; (void)ws_size;
  const float* x   = (const float*)d_in[0];
  const float* W1  = (const float*)d_in[1];
  // d_in[2] = b1 : cancels inside BN1, unused
  const float* g1  = (const float*)d_in[3];
  const float* be1 = (const float*)d_in[4];
  const float* W2  = (const float*)d_in[5];
  const float* b2  = (const float*)d_in[6];
  const float* W3  = (const float*)d_in[7];
  // d_in[8] = b3 : cancels inside BN2, unused
  const float* g2  = (const float*)d_in[9];
  const float* be2 = (const float*)d_in[10];
  const float* W4  = (const float*)d_in[11];
  const float* b4  = (const float*)d_in[12];

  float* ws    = (float*)d_ws;
  float* part1 = ws + OFF_PART1;
  float* e1s   = ws + OFF_E1S;
  float* e1b   = ws + OFF_E1B;
  float* part2 = ws + OFF_PART2;
  float* e2s   = ws + OFF_E2S;
  float* e2b   = ws + OFF_E2B;
  float* feats = ws + OFF_FEATS;
  float* nxyz  = ws + OFF_NXYZ;
  float* pool  = ws + OFF_POOL;
  float* tbuf  = ws + OFF_T;
  float* out   = (float*)d_out;

  stats_x_partial<<<64, 256, 0, stream>>>(x, part1);
  bn1_finalize<<<1, 32, 0, stream>>>(part1, W1, g1, be1, e1s, e1b);
  feats_kernel<<<NPTS / (16 * FW), 32 * FW, 0, stream>>>(x, W1, e1s, e1b, W2,
                                                         b2, feats);
  fps_kernel<<<BB, 512, 0, stream>>>(x, nxyz);
  knn_pool_kernel<<<NQ / KW, 32 * KW, 0, stream>>>(x, feats, nxyz, pool);
  mlp3_kernel<<<NQ / (16 * MW), 32 * MW, 0, stream>>>(pool, W3, tbuf);
  bn2_partial<<<128, 256, 0, stream>>>(tbuf, part2);
  bn2_finalize<<<1, 64, 0, stream>>>(part2, g2, be2, e2s, e2b);
  final_mlp_kernel<<<NQ / (16 * MW), 32 * MW, 0, stream>>>(tbuf, e2s, e2b, W4,
                                                           b4, out);
}
